// Stickbreaking_51127290692269
// MI455X (gfx1250) — compile-verified, matrix operand order checked by reading, not run on hardware
//
#include <hip/hip_runtime.h>

// Stickbreaking attention, fused flash-style, reverse key streaming.
// B=2 H=16 S=2048 D=128 fp32 in/out, bf16 WMMA compute (v_wmma_f32_16x16x32_bf16).
//
// att[i,t] = sigmoid(x_it) * PROD_{j>=t} sigmoid(-x_ij)   (product form of the
// reference's z * exp(reversed-cumsum(log_sigmoid(-x))) -- identical math, no logs).
// Suffix cum-product within a 16-key tile is done with v_permlane16_b32 (pure
// VALU); a per-row multiplicative carry handles the cross-tile part. Once the
// carry is exactly 0 for every row in the block, all remaining (lower) key
// blocks contribute exactly 0 -> uniform early exit.
// V is staged TRANSPOSED in LDS so every WMMA B-operand half is one ds_load_b128.

typedef __attribute__((ext_vector_type(16))) __bf16    v16bf;
typedef __attribute__((ext_vector_type(8)))  float     v8f;
typedef __attribute__((ext_vector_type(8)))  unsigned  v8u;

#define S_LEN 2048
#define D_DIM 128
#define KB    32      // keys processed per block iteration
#define QROWS 128     // query rows per workgroup (8 waves * 16)
#define KSTR  136     // sK row stride (u16): 128 + 8 pad (bank spread, 16B align)
#define VSTR  40      // sVt row stride (u16): 32 + 8 pad
#define ASTR  40      // sAtt row stride (u16): 32 + 8 pad

__device__ __forceinline__ unsigned short f2bfu(float f) {
    unsigned u = __builtin_bit_cast(unsigned, f);
    u += 0x7FFFu + ((u >> 16) & 1u);          // round-to-nearest-even
    return (unsigned short)(u >> 16);
}
__device__ __forceinline__ __bf16 u2bf(unsigned short s) {
    return __builtin_bit_cast(__bf16, s);
}
__device__ __forceinline__ int apat(int r) {  // A/B 16-bit WMMA K-pattern per VGPR
    return (r < 4) ? (2 * r) : (16 + 2 * (r - 4));
}

// pack two f32 -> 2x bf16 in one dword (v_cvt_pk_bf16_f32 when available)
#if __has_builtin(__builtin_amdgcn_cvt_pk_bf16_f32)
__device__ __forceinline__ unsigned pk_bf16(float a, float b) {
    return __builtin_bit_cast(unsigned, __builtin_amdgcn_cvt_pk_bf16_f32(a, b));
}
#else
__device__ __forceinline__ unsigned pk_bf16(float a, float b) {
    return (unsigned)f2bfu(a) | ((unsigned)f2bfu(b) << 16);
}
#endif

// cross-lane permute within each 16-lane half: pure VALU, no LDS round trip.
#if __has_builtin(__builtin_amdgcn_permlane16)
#define HAVE_PL16 1
__device__ __forceinline__ float pl16(float v, unsigned lo, unsigned hi) {
    const int i = __builtin_bit_cast(int, v);
    return __builtin_bit_cast(float,
        __builtin_amdgcn_permlane16(i, i, lo, hi, false, false));
}
#else
#define HAVE_PL16 0
#endif

// assemble a v16bf WMMA operand from two 16-byte LDS loads
__device__ __forceinline__ v16bf mk_op(const unsigned short* p_lo,
                                       const unsigned short* p_hi) {
    const uint4 a = *(const uint4*)p_lo;      // elems 0..7  (VGPR 0-3)
    const uint4 b = *(const uint4*)p_hi;      // elems 8..15 (VGPR 4-7)
    v8u w;
    w[0] = a.x; w[1] = a.y; w[2] = a.z; w[3] = a.w;
    w[4] = b.x; w[5] = b.y; w[6] = b.z; w[7] = b.w;
    return __builtin_bit_cast(v16bf, w);
}

__global__ __launch_bounds__(256) void sb_attn_kernel(
    const float* __restrict__ q, const float* __restrict__ k,
    const float* __restrict__ v, float* __restrict__ out)
{
    __shared__ unsigned short sK[KB * KSTR];          // 8.5 KB, bf16 [key][d]
    __shared__ unsigned short sVt[D_DIM * VSTR];      // 10 KB, bf16 [d][key] (transposed)
    __shared__ unsigned short sAtt[8][16 * ASTR];     // 10 KB, per-wave att [m][k]

    const int tid  = threadIdx.x;
    const int wave = tid >> 5;
    const int lane = tid & 31;
    const int l16  = lane & 15;
    const int hi8  = (lane >> 4) << 3;               // 0 for lanes 0-15, 8 for 16-31

    const int bh = blockIdx.y;                       // b*H + h
    const int m0 = blockIdx.x * QROWS + wave * 16;   // this wave's query row base

    const float scale = 0.08838834764831845f;        // 1/sqrt(128)

    // ---- Load Q tile (16 x 128) into WMMA A-layout, pre-scaled -------------
    v16bf qa[4];
    {
        const float* qrow = q + ((size_t)bh * S_LEN + (m0 + l16)) * D_DIM;
        #pragma unroll
        for (int c = 0; c < 4; ++c) {
            #pragma unroll
            for (int r = 0; r < 8; ++r) {
                const int d0 = c * 32 + apat(r) + hi8;
                float2 f = *(const float2*)(qrow + d0);
                qa[c][2 * r]     = u2bf(f2bfu(f.x * scale));
                qa[c][2 * r + 1] = u2bf(f2bfu(f.y * scale));
            }
        }
    }

    v8f acc[8];
    #pragma unroll
    for (int t = 0; t < 8; ++t) acc[t] = (v8f)0.0f;
    v8f carry = (v8f)1.0f;                           // per-row suffix PRODUCT carry

    const float* kbase = k + (size_t)bh * S_LEN * D_DIM;
    const float* vbase = v + (size_t)bh * S_LEN * D_DIM;

    // V staging assignment: thread owns one d column, 16 keys.
    const int vd = tid & 127;                        // d column
    const int vk = (tid >> 7) << 4;                  // key base: 0 or 16

    // ---- Stream key blocks from the END of the sequence backwards ----------
    for (int kb = S_LEN / KB - 1; kb >= 0; --kb) {
        {   // ---- stage K row-major: 32x128 f32 -> bf16, padded rows --------
            const float* kt = kbase + (size_t)kb * KB * D_DIM;
            #pragma unroll
            for (int i = tid; i < (KB * D_DIM) / 4; i += 256) {
                float4 kf = *(const float4*)(kt + 4 * i);
                const int row = i >> 5, col = (4 * i) & 127;
                *(uint2*)(&sK[row * KSTR + col]) =
                    make_uint2(pk_bf16(kf.x, kf.y), pk_bf16(kf.z, kf.w));
            }
            // ---- stage V TRANSPOSED: coalesced b32 loads, 2x b128 stores ---
            const float* vt = vbase + (size_t)kb * KB * D_DIM;
            float vv[16];
            #pragma unroll
            for (int j = 0; j < 16; ++j)
                vv[j] = vt[(size_t)(vk + j) * D_DIM + vd];
            *(uint4*)(&sVt[vd * VSTR + vk]) = make_uint4(
                pk_bf16(vv[0], vv[1]), pk_bf16(vv[2], vv[3]),
                pk_bf16(vv[4], vv[5]), pk_bf16(vv[6], vv[7]));
            *(uint4*)(&sVt[vd * VSTR + vk + 8]) = make_uint4(
                pk_bf16(vv[8], vv[9]),  pk_bf16(vv[10], vv[11]),
                pk_bf16(vv[12], vv[13]), pk_bf16(vv[14], vv[15]));

            if (kb > 0 && tid < 64) {                // prefetch next (lower) tile
                __builtin_prefetch(kt - KB * D_DIM + tid * 64, 0, 1);
                __builtin_prefetch(vt - KB * D_DIM + tid * 64, 0, 1);
            }
        }
        __syncthreads();

        // ---- two 16-key sub-tiles, HIGH keys first (reverse cum-product) ---
        #pragma unroll
        for (int sub = 1; sub >= 0; --sub) {
            const int n0 = sub * 16;

            // logits(16x16) = Q(16x128) * K^T : 4 WMMAs over the D dimension
            v8f lg = (v8f)0.0f;
            const unsigned short* kp = &sK[(n0 + l16) * KSTR];
            #pragma unroll
            for (int c = 0; c < 4; ++c) {
                const v16bf kmat = mk_op(kp + c * 32 + hi8, kp + c * 32 + 16 + hi8);
                lg = __builtin_amdgcn_wmma_f32_16x16x32_bf16(
                        false, qa[c], false, kmat, (short)0, lg, false, false);
            }

            // elementwise: z = sigmoid(x), beta = sigmoid(-x) = e^-x * z;
            // inclusive suffix PRODUCT of beta across the 16 key lanes.
            v8f attv;
            #pragma unroll
            for (int r = 0; r < 8; ++r) {
                const float x = lg[r];
                const float t = __expf(-x);
                const float z = __builtin_amdgcn_rcpf(1.0f + t);
                float s = t * z;                         // beta in (0,1)
#if HAVE_PL16
                float sh;
                sh = pl16(s, 0x87654321u, 0x0fedcba9u);  // lane i <- i+1 (mod 16)
                s *= (l16 < 15) ? sh : 1.0f;
                sh = pl16(s, 0x98765432u, 0x10fedcbau);  // lane i <- i+2
                s *= (l16 < 14) ? sh : 1.0f;
                sh = pl16(s, 0xba987654u, 0x3210fedcu);  // lane i <- i+4
                s *= (l16 < 12) ? sh : 1.0f;
                sh = pl16(s, 0xfedcba98u, 0x76543210u);  // lane i <- i+8
                s *= (l16 < 8) ? sh : 1.0f;
                const float tot = pl16(s, 0x00000000u, 0x00000000u); // key-0 bcast
#else
                #pragma unroll
                for (int off = 1; off < 16; off <<= 1) {
                    const float sh = __shfl_down(s, off, 16);
                    s *= (l16 + off < 16) ? sh : 1.0f;
                }
                const float tot = __shfl(s, lane & 16, 32);
#endif
                attv[r]  = z * s * carry[r];             // z * PROD_{j>=t} beta
                carry[r] *= tot;
            }

            // spill att sub-tile (C-layout) to LDS as bf16 in [m][k] order
            unsigned short* ap = &sAtt[wave][0];
            #pragma unroll
            for (int r = 0; r < 8; ++r)
                ap[(r + hi8) * ASTR + n0 + l16] = f2bfu(attv[r]);
        }

        // same-wave LDS store->load ordering (CDNA5 split counter) + compiler fence
        asm volatile("s_wait_dscnt 0" ::: "memory");

        // ---- reload att(16x32) in A-layout, then out += att * V -----------
        const unsigned short* ap = &sAtt[wave][l16 * ASTR];
        const v16bf amat = mk_op(ap + hi8, ap + 16 + hi8);
        #pragma unroll
        for (int t = 0; t < 8; ++t) {
            const unsigned short* vp = &sVt[(t * 16 + l16) * VSTR];
            const v16bf bmat = mk_op(vp + hi8, vp + 16 + hi8);
            acc[t] = __builtin_amdgcn_wmma_f32_16x16x32_bf16(
                        false, amat, false, bmat, (short)0, acc[t], false, false);
        }

        // ---- exact early exit: once every row's carry is 0, all remaining
        // (lower-index) key blocks contribute exactly 0. Doubles as the
        // end-of-iteration barrier protecting sK/sVt before re-staging.
        float csum = 0.0f;
        #pragma unroll
        for (int r = 0; r < 8; ++r) csum += carry[r];    // carries are >= 0
        if (__syncthreads_and(csum == 0.0f)) break;
    }

    // ---- epilogue: f32 output tile 16x128 ----------------------------------
    {
        float* orow = out + ((size_t)bh * S_LEN + m0) * D_DIM;
        #pragma unroll
        for (int t = 0; t < 8; ++t) {
            #pragma unroll
            for (int r = 0; r < 8; ++r) {
                orow[(size_t)(r + hi8) * D_DIM + t * 16 + l16] = acc[t][r];
            }
        }
    }
}

extern "C" void kernel_launch(void* const* d_in, const int* in_sizes, int n_in,
                              void* d_out, int out_size, void* d_ws, size_t ws_size,
                              hipStream_t stream) {
    const float* q = (const float*)d_in[0];
    const float* k = (const float*)d_in[1];
    const float* v = (const float*)d_in[2];
    float* out = (float*)d_out;

    const int B = 2, H = 16;
    dim3 grid(S_LEN / QROWS, B * H);   // 16 x 32 blocks
    dim3 block(256);                   // 8 waves of 32
    sb_attn_kernel<<<grid, block, 0, stream>>>(q, k, v, out);
}
